// Dynamics_75015898791961
// MI455X (gfx1250) — matrix operand order repaired and checked by reading, not compiled, not run on hardware
//
#include <hip/hip_runtime.h>
#include <hip/hip_bf16.h>

typedef __bf16 bf16_t;
typedef __attribute__((ext_vector_type(16))) __bf16 v16bf;
typedef __attribute__((ext_vector_type(8)))  float  v8f;

#define BATCH 1024
#define XD    512
#define HD    1024
#define NHEAD 1536

// ---------------------------------------------------------------- utilities

__device__ inline float wrap_angle(float x) {
  const float PIF = 3.14159265358979323846f;
  float w = x + PIF;
  w -= 6.283185307179586f * floorf(w * 0.15915494309189535f);
  return w - PIF;
}

// blockDim.x == 256 assumed
__device__ inline float block_reduce_sum(float v, float* red) {
  int tid = threadIdx.x;
  red[tid] = v;
  __syncthreads();
  for (int s = 128; s > 0; s >>= 1) {
    if (tid < s) red[tid] += red[tid + s];
    __syncthreads();
  }
  float r = red[0];
  __syncthreads();
  return r;
}

// plaquette p(t,l) for one batch row (T=L=16, 2 links/site)
__device__ inline float plaq(const float* __restrict__ xr, int t, int l) {
  t &= 15; l &= 15;
  int lp = (l + 1) & 15, tp = (t + 1) & 15;
  return xr[(t * 16 + l) * 2 + 0] - xr[(t * 16 + l) * 2 + 1]
       - xr[(t * 16 + lp) * 2 + 0] + xr[(tp * 16 + l) * 2 + 1];
}

// ---------------------------------------------------------------- WMMA GEMM
// C[M,N] = act(A[M,K] * B[K,N] + bias), B supplied pre-transposed (Bt[N][K], bf16).
// 128x128 block tile, 8 waves (wave32): each wave owns 32(M) x 64(N) = 2x4 WMMA tiles.
// K blocked by 64 through double-buffered LDS (2x16KB A + 2x16KB B = 64KB);
// two 16x16x32 bf16 WMMA K-substeps per K-block, one barrier per K-block.

// A fragment: 16x32 bf16, row-major [m][k] in LDS (stride 64).
// ISA layout: lanes 0-15 = rows, half selects K interleave:
//   VGPR v<4 : K = h*8 + 2v..2v+1 ; VGPR v>=4 : K = 16 + h*8 + 2(v-4)..
__device__ inline v16bf lds_frag_a(const bf16_t* s, int row0, int ks, int lane) {
  const int m = row0 + (lane & 15);
  const int h = (lane >> 4) & 1;
  const bf16_t* p = s + m * 64 + ks;
  v16bf f;
#pragma unroll
  for (int v = 0; v < 8; ++v) {
    int kk = ((v & 4) << 2) + h * 8 + (v & 3) * 2;
    f[2 * v]     = p[kk];
    f[2 * v + 1] = p[kk + 1];
  }
  return f;
}

// B fragment: 32x16 bf16 fed from Bt[n][k] row-major in LDS (stride 64).
// ISA layout: same K interleave as A: VGPR v<4 : K = h*8 + 2v..2v+1 ; VGPR v>=4 : K = 16 + h*8 + 2(v-4)..
__device__ inline v16bf lds_frag_b(const bf16_t* s, int row0, int ks, int lane) {
  const int n = row0 + (lane & 15);
  const int h = (lane >> 4) & 1;
  const bf16_t* p = s + n * 64 + ks + h * 8;
  v16bf f;
#pragma unroll
  for (int v = 0; v < 8; ++v) {
    f[2 * v]     = p[2 * v + ((v & 4) << 1)];
    f[2 * v + 1] = p[2 * v + ((v & 4) << 1) + 1];
  }
  return f;
}

// act == 0: tanh, write bf16 to outb (feeds next layer)
// act == 1: heads S|T|Q over N=1536 -> tanh on cols [0,512) and [1024,1536), write f32
__global__ __launch_bounds__(256) void wmma_gemm_kernel(
    const bf16_t* __restrict__ A, const bf16_t* __restrict__ Bt,
    const float* __restrict__ bias, float* __restrict__ outf,
    bf16_t* __restrict__ outb, int M, int N, int K, int act) {
  __shared__ bf16_t sA[2][128 * 64];
  __shared__ bf16_t sB[2][128 * 64];
  const int tid  = threadIdx.x;
  const int lane = tid & 31;
  const int wid  = tid >> 5;
  const int wm   = (wid & 3) * 32;  // wave M offset inside block tile
  const int wn   = (wid >> 2) * 64; // wave N offset inside block tile
  const int m0   = blockIdx.y * 128;
  const int n0   = blockIdx.x * 128;

  v8f acc[2][4] = {};

  // stage 128x64 bf16 tiles of A and Bt into LDS buffer `buf` (uint4 = 8 elems)
  auto stage = [&](int buf, int k0) {
    for (int i = tid; i < 1024; i += 256) {
      int r = i >> 3, cg = (i & 7) << 3;
      *reinterpret_cast<uint4*>(sA[buf] + r * 64 + cg) =
          *reinterpret_cast<const uint4*>(A + (size_t)(m0 + r) * K + k0 + cg);
      *reinterpret_cast<uint4*>(sB[buf] + r * 64 + cg) =
          *reinterpret_cast<const uint4*>(Bt + (size_t)(n0 + r) * K + k0 + cg);
    }
  };

  stage(0, 0);
  __syncthreads();

  const int nkb = K >> 6;
  for (int kb = 0; kb < nkb; ++kb) {
    const int cur = kb & 1;
    // issue next K-block's global loads first so they overlap the WMMAs below
    if (kb + 1 < nkb) stage(cur ^ 1, (kb + 1) << 6);
    if (kb + 2 < nkb) {  // pull K+128 block toward L2 early (global_prefetch_b8)
      __builtin_prefetch(A  + (size_t)(m0 + (tid >> 1)) * K + ((kb + 2) << 6) + (tid & 1) * 32, 0, 0);
      __builtin_prefetch(Bt + (size_t)(n0 + (tid >> 1)) * K + ((kb + 2) << 6) + (tid & 1) * 32, 0, 0);
    }
#pragma unroll
    for (int ks = 0; ks < 64; ks += 32) {
      v16bf af[2], bfr[4];
#pragma unroll
      for (int i = 0; i < 2; ++i) af[i] = lds_frag_a(sA[cur], wm + i * 16, ks, lane);
#pragma unroll
      for (int j = 0; j < 4; ++j) bfr[j] = lds_frag_b(sB[cur], wn + j * 16, ks, lane);
#pragma unroll
      for (int i = 0; i < 2; ++i)
#pragma unroll
        for (int j = 0; j < 4; ++j)
          acc[i][j] = __builtin_amdgcn_wmma_f32_16x16x32_bf16(
              false, af[i], false, bfr[j], (short)0, acc[i][j], false, false);
    }
    // one barrier per K-block: reads of `cur` done, writes to `cur^1` visible
    __syncthreads();
  }

  // epilogue: C/D layout -> VGPR r holds (row = r + 8*half, col = lane&15)
  const int cn = lane & 15;
  const int hh = (lane >> 4) & 1;
#pragma unroll
  for (int i = 0; i < 2; ++i) {
#pragma unroll
    for (int j = 0; j < 4; ++j) {
      int col  = n0 + wn + j * 16 + cn;
      float bv = bias ? bias[col] : 0.0f;
#pragma unroll
      for (int r = 0; r < 8; ++r) {
        int row = m0 + wm + i * 16 + hh * 8 + r;
        float v = acc[i][j][r] + bv;
        if (act == 0) {
          v = tanhf(v);
          outb[(size_t)row * N + col] = (bf16_t)v;
        } else {
          if (col < 512 || col >= 1024) v = tanhf(v);
          outf[(size_t)row * N + col] = v;
        }
      }
    }
  }
}

// ---------------------------------------------------------------- packing

// W[K][N] f32 -> Wt[N][K] bf16
__global__ void pack_weight_kernel(const float* __restrict__ W, bf16_t* __restrict__ Wt,
                                   int K, int N) {
  int idx = blockIdx.x * blockDim.x + threadIdx.x;
  if (idx >= K * N) return;
  int k = idx / N, n = idx % N;
  Wt[(size_t)n * K + k] = (bf16_t)W[idx];
}

// out[b][0..511] = a[b]; out[b][512..1023] = (mask? m*bsrc : bsrc)[b]  (bf16)
__global__ void build_input_kernel(const float* __restrict__ a, const float* __restrict__ bsrc,
                                   const float* __restrict__ mask, int comp,
                                   bf16_t* __restrict__ out) {
  int idx = blockIdx.x * blockDim.x + threadIdx.x;  // BATCH*1024
  int i = idx & 1023, bb = idx >> 10;
  float v;
  if (i < 512) {
    v = a[bb * 512 + i];
  } else {
    int j = i - 512;
    v = bsrc[bb * 512 + j];
    if (mask) { float m = mask[j]; if (comp) m = 1.0f - m; v *= m; }
  }
  out[idx] = (bf16_t)v;
}

// ---------------------------------------------------------------- physics

__global__ __launch_bounds__(256) void grad_action_kernel(
    const float* __restrict__ x, float* __restrict__ g, const float* __restrict__ beta_p) {
  int b = blockIdx.x;
  int t = threadIdx.x >> 4, l = threadIdx.x & 15;
  const float* xr = x + b * 512;
  float beta = beta_p[0];
  float sp  = sinf(plaq(xr, t, l));
  float spl = sinf(plaq(xr, t, l - 1));
  float spt = sinf(plaq(xr, t - 1, l));
  g[b * 512 + (t * 16 + l) * 2 + 0] = beta * (sp - spl);
  g[b * 512 + (t * 16 + l) * 2 + 1] = beta * (spt - sp);
}

__global__ __launch_bounds__(256) void action_kernel(
    const float* __restrict__ x, float* __restrict__ out, const float* __restrict__ beta_p) {
  __shared__ float red[256];
  int b = blockIdx.x;
  int t = threadIdx.x >> 4, l = threadIdx.x & 15;
  float p = plaq(x + b * 512, t, l);
  float s = block_reduce_sum(1.0f - cosf(p), red);
  if (threadIdx.x == 0) out[b] = beta_p[0] * s;
}

// ---------------------------------------------------------------- updates

// dir 0 = forward, 1 = backward
__global__ __launch_bounds__(256) void update_v_kernel(
    float* __restrict__ v, const float* __restrict__ g, const float* __restrict__ heads,
    const float* __restrict__ eps_p, float* __restrict__ ld, int dir) {
  __shared__ float red[256];
  int b = blockIdx.x;
  float eps = eps_p[0];
  float acc = 0.0f;
  for (int e = threadIdx.x; e < 512; e += 256) {
    float S  = heads[b * NHEAD + e];
    float Tt = heads[b * NHEAD + 512 + e];
    float Q  = heads[b * NHEAD + 1024 + e];
    float vv = v[b * 512 + e];
    float gg = g[b * 512 + e];
    float geT = gg * expf(eps * Q) + Tt;
    if (dir == 0) {
      float sc = 0.5f * eps * S;
      vv = vv * expf(sc) - 0.5f * eps * geT;
      acc += sc;
    } else {
      float sc = -0.5f * eps * S;
      vv = expf(sc) * (vv + 0.5f * eps * geT);
      acc += sc;
    }
    v[b * 512 + e] = vv;
  }
  float s = block_reduce_sum(acc, red);
  if (threadIdx.x == 0) ld[b] += s;
}

__global__ __launch_bounds__(256) void update_x_kernel(
    float* __restrict__ x, const float* __restrict__ v, const float* __restrict__ heads,
    const float* __restrict__ mask, int comp, const float* __restrict__ eps_p,
    float* __restrict__ ld, int dir) {
  __shared__ float red[256];
  int b = blockIdx.x;
  float eps = eps_p[0];
  float acc = 0.0f;
  for (int e = threadIdx.x; e < 512; e += 256) {
    float m = mask[e]; if (comp) m = 1.0f - m;
    float mb = 1.0f - m;
    float S  = heads[b * NHEAD + e];
    float Tt = heads[b * NHEAD + 512 + e];
    float Q  = heads[b * NHEAD + 1024 + e];
    float xx = x[b * 512 + e];
    float vv = v[b * 512 + e];
    float veT = vv * expf(eps * Q) + Tt;
    float xn;
    if (dir == 0) {
      float sc = eps * S;
      xn = m * xx + mb * (xx * expf(sc) + eps * veT);
      acc += mb * sc;
    } else {
      float sc = -eps * S;
      xn = m * xx + mb * expf(sc) * (xx - eps * veT);
      acc += mb * sc;
    }
    x[b * 512 + e] = wrap_angle(xn);
  }
  float s = block_reduce_sum(acc, red);
  if (threadIdx.x == 0) ld[b] += s;
}

// ---------------------------------------------------------------- finalize

__global__ void select_kernel(const float* __restrict__ xf, const float* __restrict__ xb,
                              const float* __restrict__ vf, const float* __restrict__ vb,
                              const float* __restrict__ ldf, const float* __restrict__ ldb,
                              const float* __restrict__ rdir,
                              float* __restrict__ x_prop, float* __restrict__ v_prop,
                              float* __restrict__ sld_prop) {
  int idx = blockIdx.x * blockDim.x + threadIdx.x;  // BATCH*512
  int b = idx >> 9;
  bool f = rdir[b] < 0.5f;
  x_prop[idx] = f ? xf[idx] : xb[idx];
  v_prop[idx] = f ? vf[idx] : vb[idx];
  if ((idx & 511) == 0) sld_prop[b] = f ? ldf[b] : ldb[b];
}

__global__ __launch_bounds__(256) void accept_kernel(
    const float* __restrict__ x, const float* __restrict__ v_in,
    const float* __restrict__ x_prop, const float* __restrict__ v_prop,
    const float* __restrict__ sld_prop, const float* __restrict__ act_init,
    const float* __restrict__ act_prop, const float* __restrict__ rdir,
    const float* __restrict__ racc, float* __restrict__ out) {
  __shared__ float red[256];
  __shared__ float sh_ma, sh_px;
  int b = blockIdx.x;
  bool f = rdir[b] < 0.5f;
  const float* vi = v_in + (f ? (size_t)0 : (size_t)BATCH * 512) + (size_t)b * 512;
  float a = 0.0f, c = 0.0f;
  for (int e = threadIdx.x; e < 512; e += 256) {
    float q = vi[e];               a += q * q;
    float r = v_prop[b * 512 + e]; c += r * r;
  }
  float sa = block_reduce_sum(a, red);
  float sc = block_reduce_sum(c, red);
  if (threadIdx.x == 0) {
    float h_init = act_init[b] + 0.5f * sa;
    float h_prop = act_prop[b] + 0.5f * sc;
    float px = expf(fminf(h_init - h_prop + sld_prop[b], 0.0f));
    sh_px = px;
    sh_ma = (racc[b] < px) ? 1.0f : 0.0f;
  }
  __syncthreads();
  float ma = sh_ma;
  for (int e = threadIdx.x; e < 512; e += 256) {
    float xo = ma * x_prop[b * 512 + e] + (1.0f - ma) * x[b * 512 + e];
    out[(size_t)b * 512 + e] = wrap_angle(xo);
  }
  if (threadIdx.x == 0) {
    out[(size_t)BATCH * 512 + b] = sh_px;
    out[(size_t)BATCH * 512 + BATCH + b] = sh_ma * sld_prop[b];
  }
}

// ---------------------------------------------------------------- host side

extern "C" void kernel_launch(void* const* d_in, const int* in_sizes, int n_in,
                              void* d_out, int out_size, void* d_ws, size_t ws_size,
                              hipStream_t stream) {
  (void)in_sizes; (void)n_in; (void)out_size; (void)ws_size;

  const float* x_in   = (const float*)d_in[0];
  const float* v_in   = (const float*)d_in[1];
  const float* eps_p  = (const float*)d_in[2];
  const float* beta_p = (const float*)d_in[3];
  const float* rdir   = (const float*)d_in[4];
  const float* racc   = (const float*)d_in[5];
  const float* masks  = (const float*)d_in[6];
  // order per net: W1 b1 W2 b2 Ws bs Wt bt Wq bq
  const float* xW[5] = {(const float*)d_in[7],  (const float*)d_in[9],
                        (const float*)d_in[11], (const float*)d_in[13], (const float*)d_in[15]};
  const float* xb[5] = {(const float*)d_in[8],  (const float*)d_in[10],
                        (const float*)d_in[12], (const float*)d_in[14], (const float*)d_in[16]};
  const float* vW[5] = {(const float*)d_in[17], (const float*)d_in[19],
                        (const float*)d_in[21], (const float*)d_in[23], (const float*)d_in[25]};
  const float* vb[5] = {(const float*)d_in[18], (const float*)d_in[20],
                        (const float*)d_in[22], (const float*)d_in[24], (const float*)d_in[26]};

  // workspace layout
  char* ws = (char*)d_ws;
  size_t off = 0;
  auto walloc = [&](size_t bytes) {
    void* p = ws + off;
    off = (off + bytes + 255) & ~(size_t)255;
    return p;
  };
  bf16_t* xW1t = (bf16_t*)walloc((size_t)HD * HD * 2);
  bf16_t* xW2t = (bf16_t*)walloc((size_t)HD * HD * 2);
  bf16_t* xWht = (bf16_t*)walloc((size_t)NHEAD * HD * 2);
  bf16_t* vW1t = (bf16_t*)walloc((size_t)HD * HD * 2);
  bf16_t* vW2t = (bf16_t*)walloc((size_t)HD * HD * 2);
  bf16_t* vWht = (bf16_t*)walloc((size_t)NHEAD * HD * 2);
  float*  xbh  = (float*)walloc(NHEAD * 4);
  float*  vbh  = (float*)walloc(NHEAD * 4);
  bf16_t* Ain  = (bf16_t*)walloc((size_t)BATCH * 1024 * 2);
  bf16_t* h1b  = (bf16_t*)walloc((size_t)BATCH * HD * 2);
  bf16_t* h2b  = (bf16_t*)walloc((size_t)BATCH * HD * 2);
  float*  heads = (float*)walloc((size_t)BATCH * NHEAD * 4);
  float*  g    = (float*)walloc((size_t)BATCH * XD * 4);
  float*  xF   = (float*)walloc((size_t)BATCH * XD * 4);
  float*  vF   = (float*)walloc((size_t)BATCH * XD * 4);
  float*  xB_  = (float*)walloc((size_t)BATCH * XD * 4);
  float*  vB_  = (float*)walloc((size_t)BATCH * XD * 4);
  float*  ldF  = (float*)walloc(BATCH * 4);
  float*  ldB  = (float*)walloc(BATCH * 4);
  float*  x_prop = (float*)walloc((size_t)BATCH * XD * 4);
  float*  v_prop = (float*)walloc((size_t)BATCH * XD * 4);
  float*  sld_prop = (float*)walloc(BATCH * 4);
  float*  act_init = (float*)walloc(BATCH * 4);
  float*  act_prop = (float*)walloc(BATCH * 4);

  const int TPB = 256;

  // ---- pack weights (transposed bf16), pack head biases
  pack_weight_kernel<<<(HD * HD + TPB - 1) / TPB, TPB, 0, stream>>>(xW[0], xW1t, HD, HD);
  pack_weight_kernel<<<(HD * HD + TPB - 1) / TPB, TPB, 0, stream>>>(xW[1], xW2t, HD, HD);
  pack_weight_kernel<<<(HD * HD + TPB - 1) / TPB, TPB, 0, stream>>>(vW[0], vW1t, HD, HD);
  pack_weight_kernel<<<(HD * HD + TPB - 1) / TPB, TPB, 0, stream>>>(vW[1], vW2t, HD, HD);
  for (int p = 0; p < 3; ++p) {
    pack_weight_kernel<<<(HD * XD + TPB - 1) / TPB, TPB, 0, stream>>>(
        xW[2 + p], xWht + (size_t)p * XD * HD, HD, XD);
    pack_weight_kernel<<<(HD * XD + TPB - 1) / TPB, TPB, 0, stream>>>(
        vW[2 + p], vWht + (size_t)p * XD * HD, HD, XD);
    hipMemcpyAsync(xbh + p * XD, xb[2 + p], XD * 4, hipMemcpyDeviceToDevice, stream);
    hipMemcpyAsync(vbh + p * XD, vb[2 + p], XD * 4, hipMemcpyDeviceToDevice, stream);
  }

  // ---- init chain state
  hipMemsetAsync(ldF, 0, BATCH * 4, stream);
  hipMemsetAsync(ldB, 0, BATCH * 4, stream);
  hipMemcpyAsync(xF,  x_in, (size_t)BATCH * XD * 4, hipMemcpyDeviceToDevice, stream);
  hipMemcpyAsync(xB_, x_in, (size_t)BATCH * XD * 4, hipMemcpyDeviceToDevice, stream);
  hipMemcpyAsync(vF,  v_in, (size_t)BATCH * XD * 4, hipMemcpyDeviceToDevice, stream);
  hipMemcpyAsync(vB_, v_in + (size_t)BATCH * XD, (size_t)BATCH * XD * 4,
                 hipMemcpyDeviceToDevice, stream);

  auto run_net = [&](const bf16_t* W1t, const float* b1, const bf16_t* W2t, const float* b2,
                     const bf16_t* Wht, const float* bh) {
    wmma_gemm_kernel<<<dim3(8, 8), TPB, 0, stream>>>(Ain, W1t, b1, nullptr, h1b,
                                                     BATCH, HD, 1024, 0);
    wmma_gemm_kernel<<<dim3(8, 8), TPB, 0, stream>>>(h1b, W2t, b2, nullptr, h2b,
                                                     BATCH, HD, HD, 0);
    wmma_gemm_kernel<<<dim3(12, 8), TPB, 0, stream>>>(h2b, Wht, bh, heads, nullptr,
                                                      BATCH, NHEAD, HD, 1);
  };

  auto do_update_v = [&](float* x, float* v, float* ld, int dir) {
    grad_action_kernel<<<BATCH, TPB, 0, stream>>>(x, g, beta_p);
    build_input_kernel<<<(BATCH * 1024) / TPB, TPB, 0, stream>>>(x, g, nullptr, 0, Ain);
    run_net(vW1t, vb[0], vW2t, vb[1], vWht, vbh);
    update_v_kernel<<<BATCH, TPB, 0, stream>>>(v, g, heads, eps_p, ld, dir);
  };

  auto do_update_x = [&](float* x, float* v, float* ld, const float* m, int comp, int dir) {
    build_input_kernel<<<(BATCH * 1024) / TPB, TPB, 0, stream>>>(v, x, m, comp, Ain);
    run_net(xW1t, xb[0], xW2t, xb[1], xWht, xbh);
    update_x_kernel<<<BATCH, TPB, 0, stream>>>(x, v, heads, m, comp, eps_p, ld, dir);
  };

  // ---- forward chain
  for (int s = 0; s < 10; ++s) {
    const float* m = masks + s * XD;
    do_update_v(xF, vF, ldF, 0);
    do_update_x(xF, vF, ldF, m, 0, 0);
    do_update_x(xF, vF, ldF, m, 1, 0);
    do_update_v(xF, vF, ldF, 0);
  }
  // ---- backward chain (flipped masks; first x-update uses 1-m)
  for (int s = 0; s < 10; ++s) {
    const float* m = masks + (9 - s) * XD;
    do_update_v(xB_, vB_, ldB, 1);
    do_update_x(xB_, vB_, ldB, m, 1, 1);
    do_update_x(xB_, vB_, ldB, m, 0, 1);
    do_update_v(xB_, vB_, ldB, 1);
  }

  // ---- finalize: direction select, actions, Metropolis accept
  select_kernel<<<(BATCH * XD) / TPB, TPB, 0, stream>>>(xF, xB_, vF, vB_, ldF, ldB, rdir,
                                                        x_prop, v_prop, sld_prop);
  action_kernel<<<BATCH, TPB, 0, stream>>>(x_in, act_init, beta_p);
  action_kernel<<<BATCH, TPB, 0, stream>>>(x_prop, act_prop, beta_p);
  accept_kernel<<<BATCH, TPB, 0, stream>>>(x_in, v_in, x_prop, v_prop, sld_prop,
                                           act_init, act_prop, rdir, racc, (float*)d_out);
}
